// EnhancedChunkLayer_63917703299650
// MI455X (gfx1250) — matrix operand order repaired and checked by reading, not compiled, not run on hardware
//
#include <hip/hip_runtime.h>
#include <math.h>

#define Dm 1536
#define Sq 1024
#define Bb 8
#define Hh 12
#define DH 128
#define MAXC 256
#define MAXSEQ 1024

typedef __attribute__((ext_vector_type(16))) __bf16 v16bf;
typedef __attribute__((ext_vector_type(8)))  float  v8f;
typedef __attribute__((ext_vector_type(8)))  unsigned short u16x8;
typedef __attribute__((ext_vector_type(4)))  unsigned short u16x4;

union BF16x16 { v16bf v; unsigned short u[16]; u16x8 o[2]; };

static __device__ __forceinline__ unsigned short f2bf(float f) {
    union { float f; unsigned int u; } c; c.f = f;
    unsigned int u = c.u;
    unsigned int r = u + 0x7FFFu + ((u >> 16) & 1u);   // round-to-nearest-even
    return (unsigned short)(r >> 16);
}
static __device__ __forceinline__ float bf2f(unsigned short u) {
    union { unsigned int i; float f; } c; c.i = ((unsigned int)u) << 16;
    return c.f;
}
static __device__ __forceinline__ float gelu_exact(float x) {
    return 0.5f * x * (1.0f + erff(x * 0.7071067811865476f));
}

// ---------------------------------------------------------------------------
// Kernel 0: fp32 -> bf16 conversion (bandwidth-bound pre-pass; n % 4 == 0)
// ---------------------------------------------------------------------------
__global__ void cvt_bf16(const float* __restrict__ in, unsigned short* __restrict__ outp,
                         int nq) {
    int i = blockIdx.x * blockDim.x + threadIdx.x;
    if (i >= nq) return;
    float4 f = *(const float4*)(in + (size_t)i * 4);
    u16x4 o;
    o[0] = f2bf(f.x); o[1] = f2bf(f.y); o[2] = f2bf(f.z); o[3] = f2bf(f.w);
    *(u16x4*)(outp + (size_t)i * 4) = o;
}

// ---------------------------------------------------------------------------
// Kernel 1: segment scan per batch row (serial; 1024 iters, negligible cost)
// ---------------------------------------------------------------------------
__global__ void seg_kernel(const float* __restrict__ bnd, int* __restrict__ seg,
                           int* __restrict__ start, int* __restrict__ nch) {
    int b = blockIdx.x;
    if (threadIdx.x != 0) return;
    int cnt = 0;
    for (int s = 0; s < Sq; ++s) {
        int ib = bnd[b * Sq + s] > 0.85f;
        if (ib) { start[b * (Sq + 1) + cnt] = s; cnt++; }
        seg[b * Sq + s] = cnt - 1;
    }
    start[b * (Sq + 1) + cnt] = Sq;
    nch[b] = cnt;
}

// ---------------------------------------------------------------------------
// Kernel 2: C[M,N] = act(A[M,K] @ W[N,K]^T + bias); A,W bf16; C fp32 or bf16.
// Wave tile 32x64 (2x4 frags, 8 wmma / k=32). 4 waves stacked in M -> 128x64
// block tile; all waves share B frags via L0. All dims % 128/64 == 0.
// ---------------------------------------------------------------------------
__global__ __launch_bounds__(128)
void gemm_bf16(const unsigned short* __restrict__ A, const unsigned short* __restrict__ W,
               const float* __restrict__ bias, void* __restrict__ Cv,
               int M, int N, int K, int act, int out_bf16) {
    const int lane = threadIdx.x & 31;
    const int wave = threadIdx.x >> 5;
    const int mt = blockIdx.y * 128 + wave * 32;
    const int nt = blockIdx.x * 64;
    const int l15 = lane & 15;
    const int hiA = (lane >> 4) << 3;   // 0 or 8
    const int hiB = (lane >> 4) << 3;   // 0 or 8

    v8f acc[2][4];
    #pragma unroll
    for (int i = 0; i < 2; ++i)
        #pragma unroll
        for (int j = 0; j < 4; ++j)
            #pragma unroll
            for (int e = 0; e < 8; ++e) acc[i][j][e] = 0.0f;

    for (int kb = 0; kb < K; kb += 32) {
        BF16x16 af[2], bf[4];
        #pragma unroll
        for (int i = 0; i < 2; ++i) {
            const unsigned short* ap = A + (size_t)(mt + i * 16 + l15) * K + kb + hiA;
            af[i].o[0] = *(const u16x8*)(ap);        // halves 0..7
            af[i].o[1] = *(const u16x8*)(ap + 16);   // halves 8..15
        }
        #pragma unroll
        for (int j = 0; j < 4; ++j) {
            const unsigned short* bp = W + (size_t)(nt + j * 16 + l15) * K + kb + hiB;
            bf[j].o[0] = *(const u16x8*)(bp);
            bf[j].o[1] = *(const u16x8*)(bp + 16);
        }
        #pragma unroll
        for (int i = 0; i < 2; ++i)
            #pragma unroll
            for (int j = 0; j < 4; ++j)
                acc[i][j] = __builtin_amdgcn_wmma_f32_16x16x32_bf16(
                    false, af[i].v, false, bf[j].v, (short)0, acc[i][j], false, false);
    }

    #pragma unroll
    for (int i = 0; i < 2; ++i)
        #pragma unroll
        for (int j = 0; j < 4; ++j) {
            const int col = nt + j * 16 + l15;
            const float bs = bias[col];
            #pragma unroll
            for (int e = 0; e < 8; ++e) {
                const int row = mt + i * 16 + hiA + e;
                float v = acc[i][j][e] + bs;
                if (act) v = gelu_exact(v);
                if (out_bf16)
                    ((unsigned short*)Cv)[(size_t)row * N + col] = f2bf(v);
                else
                    ((float*)Cv)[(size_t)row * N + col] = v;
            }
        }
}

// ---------------------------------------------------------------------------
// Kernel 3: block-diagonal (per-chunk) attention, flash-style online softmax.
// Grid (S/16, H, B), 1 wave/block. qkv is bf16; QK^T via 4x wmma (dh=128 as
// 4 k=32 slices), 1/sqrt(dh) folded in post-WMMA. Key range restricted to the
// chunks spanned by the 16 query rows. ctx output: bf16 [B,S,D].
// ---------------------------------------------------------------------------
__global__ __launch_bounds__(32)
void attn_kernel(const unsigned short* __restrict__ qkv, const int* __restrict__ seg,
                 const int* __restrict__ start, unsigned short* __restrict__ ctx) {
    const int b = blockIdx.z, hd = blockIdx.y, qt = blockIdx.x;
    const int s0 = qt * 16;
    const int lane = threadIdx.x;
    const int l15 = lane & 15;
    const int hi = lane >> 4;           // 0/1 half of wave
    const int segB = b * Sq;
    const int stB = b * (Sq + 1);

    __shared__ float Plds[16][17];
    __shared__ float rowS[16];
    __shared__ float rowL[16];

    const int smax = seg[segB + s0 + 15];
    unsigned short* outp = ctx + (size_t)(b * Sq + s0) * Dm + hd * DH + lane * 4;
    if (smax < 0) {                      // whole tile before first boundary: dropped later
        u16x4 z = {0, 0, 0, 0};
        for (int m = 0; m < 16; ++m) *(u16x4*)(outp + (size_t)m * Dm) = z;
        return;
    }
    int smin = seg[segB + s0]; if (smin < 0) smin = 0;
    const int kstart = start[stB + smin];
    const int kend   = start[stB + smax + 1];

    int sq[8];
    #pragma unroll
    for (int i = 0; i < 8; ++i) sq[i] = seg[segB + s0 + hi * 8 + i];

    // Q fragments (A layout), raw bf16 loads
    BF16x16 qf[4];
    {
        const unsigned short* q0 = qkv + (size_t)(b * Sq + s0 + l15) * (3 * Dm) + hd * DH;
        #pragma unroll
        for (int sl = 0; sl < 4; ++sl) {
            const unsigned short* qp = q0 + sl * 32 + hi * 8;
            qf[sl].o[0] = *(const u16x8*)(qp);
            qf[sl].o[1] = *(const u16x8*)(qp + 16);
        }
    }

    float mrow[8], lrow[8];
    #pragma unroll
    for (int i = 0; i < 8; ++i) { mrow[i] = -1e30f; lrow[i] = 0.0f; }
    float4 accv[16];
    #pragma unroll
    for (int m = 0; m < 16; ++m) accv[m] = make_float4(0.f, 0.f, 0.f, 0.f);

    for (int kt = kstart; kt < kend; kt += 16) {
        // ---- scores = (Q @ K^T) * rsqrt(dh)  (C layout) ----
        const int kc = kt + l15;                        // this lane's key column
        const int kcc = kc < Sq ? kc : Sq - 1;
        const unsigned short* kp0 = qkv + (size_t)(b * Sq + kcc) * (3 * Dm) + Dm + hd * DH;
        v8f sacc;
        #pragma unroll
        for (int e = 0; e < 8; ++e) sacc[e] = 0.0f;
        #pragma unroll
        for (int sl = 0; sl < 4; ++sl) {
            BF16x16 kf;
            const unsigned short* kp = kp0 + sl * 32 + hi * 8;
            kf.o[0] = *(const u16x8*)(kp);
            kf.o[1] = *(const u16x8*)(kp + 16);
            sacc = __builtin_amdgcn_wmma_f32_16x16x32_bf16(
                false, qf[sl].v, false, kf.v, (short)0, sacc, false, false);
        }

        // ---- mask + online softmax (row reductions across 16-lane halves) ----
        const int sk = (kc < Sq) ? seg[segB + kc] : -0x40000000;
        #pragma unroll
        for (int i = 0; i < 8; ++i) {
            const bool msk = (sq[i] >= 0) && (sq[i] == sk);
            const float s = sacc[i] * 0.08838834764831845f;   // 128^-0.5
            float rm = msk ? s : -1e30f;
            rm = fmaxf(rm, __shfl_xor(rm, 1));
            rm = fmaxf(rm, __shfl_xor(rm, 2));
            rm = fmaxf(rm, __shfl_xor(rm, 4));
            rm = fmaxf(rm, __shfl_xor(rm, 8));
            const float mn = fmaxf(mrow[i], rm);
            float p = msk ? __expf(s - mn) : 0.0f;
            float rs = p;
            rs += __shfl_xor(rs, 1);
            rs += __shfl_xor(rs, 2);
            rs += __shfl_xor(rs, 4);
            rs += __shfl_xor(rs, 8);
            const float sc = __expf(mrow[i] - mn);
            lrow[i] = lrow[i] * sc + rs;
            mrow[i] = mn;
            Plds[hi * 8 + i][l15] = p;
            if (l15 == 0) rowS[hi * 8 + i] = sc;
        }
        __syncthreads();

        // ---- rescale accumulators, accumulate P @ V (lane owns 4 dh columns) ----
        #pragma unroll
        for (int m = 0; m < 16; ++m) {
            const float sc = rowS[m];
            accv[m].x *= sc; accv[m].y *= sc; accv[m].z *= sc; accv[m].w *= sc;
        }
        const int klim = (Sq - kt) < 16 ? (Sq - kt) : 16;
        for (int kk = 0; kk < klim; ++kk) {
            u16x4 vr = *(const u16x4*)(qkv + (size_t)(b * Sq + kt + kk) * (3 * Dm)
                                       + 2 * Dm + hd * DH + lane * 4);
            const float vx = bf2f(vr[0]), vy = bf2f(vr[1]),
                        vz = bf2f(vr[2]), vw = bf2f(vr[3]);
            #pragma unroll
            for (int m = 0; m < 16; ++m) {
                const float p = Plds[m][kk];
                accv[m].x = fmaf(p, vx, accv[m].x);
                accv[m].y = fmaf(p, vy, accv[m].y);
                accv[m].z = fmaf(p, vz, accv[m].z);
                accv[m].w = fmaf(p, vw, accv[m].w);
            }
        }
        __syncthreads();
    }

    if (l15 == 0) {
        #pragma unroll
        for (int i = 0; i < 8; ++i) rowL[hi * 8 + i] = lrow[i];
    }
    __syncthreads();
    #pragma unroll
    for (int m = 0; m < 16; ++m) {
        const float l = rowL[m];
        const float inv = l > 0.0f ? 1.0f / l : 0.0f;
        u16x4 o;
        o[0] = f2bf(accv[m].x * inv); o[1] = f2bf(accv[m].y * inv);
        o[2] = f2bf(accv[m].z * inv); o[3] = f2bf(accv[m].w * inv);
        *(u16x4*)(outp + (size_t)m * Dm) = o;
    }
}

// ---------------------------------------------------------------------------
// Kernel 4: ragged mean-pool per chunk + size embedding + positional encoding
// Reads attn fp32, writes bf16 chunks (MLP1 input).
// ---------------------------------------------------------------------------
__global__ void pool_kernel(const float* __restrict__ attn, const int* __restrict__ start,
                            const int* __restrict__ nchp, const float* __restrict__ size_emb,
                            const float* __restrict__ pos_enc,
                            unsigned short* __restrict__ chunks) {
    const int c = blockIdx.x, b = blockIdx.y;
    const int nch = nchp[b];
    int st = 0, en = 0;
    if (c < nch) { st = start[b * (Sq + 1) + c]; en = start[b * (Sq + 1) + c + 1]; }
    const int len = en - st;
    const float inv = len > 0 ? 1.0f / (float)len : 0.0f;
    const int sidx = len < (MAXSEQ - 1) ? len : (MAXSEQ - 1);
    for (int d = threadIdx.x; d < Dm; d += blockDim.x) {
        float sum = 0.0f;
        for (int s = st; s < en; ++s) sum += attn[(size_t)(b * Sq + s) * Dm + d];
        float v = sum * inv;
        if (len > 0) v += size_emb[(size_t)sidx * Dm + d];
        v += pos_enc[(size_t)c * Dm + d];
        chunks[((size_t)b * MAXC + c) * Dm + d] = f2bf(v);
    }
}

// ---------------------------------------------------------------------------
// Kernel 5: LayerNorm over D=1536 per row
// ---------------------------------------------------------------------------
__global__ void ln_kernel(const float* __restrict__ h, const float* __restrict__ g,
                          const float* __restrict__ be, float* __restrict__ out) {
    const int row = blockIdx.x;
    const float* hp = h + (size_t)row * Dm;
    __shared__ float red[256];
    const int tid = threadIdx.x;

    float s = 0.0f;
    for (int d = tid; d < Dm; d += 256) s += hp[d];
    red[tid] = s; __syncthreads();
    for (int off = 128; off > 0; off >>= 1) {
        if (tid < off) red[tid] += red[tid + off];
        __syncthreads();
    }
    const float mu = red[0] * (1.0f / Dm);
    __syncthreads();

    float v = 0.0f;
    for (int d = tid; d < Dm; d += 256) { float dv = hp[d] - mu; v += dv * dv; }
    red[tid] = v; __syncthreads();
    for (int off = 128; off > 0; off >>= 1) {
        if (tid < off) red[tid] += red[tid + off];
        __syncthreads();
    }
    const float rs = rsqrtf(red[0] * (1.0f / Dm) + 1e-5f);
    for (int d = tid; d < Dm; d += 256)
        out[(size_t)row * Dm + d] = (hp[d] - mu) * rs * g[d] + be[d];
}

// ---------------------------------------------------------------------------
extern "C" void kernel_launch(void* const* d_in, const int* in_sizes, int n_in,
                              void* d_out, int out_size, void* d_ws, size_t ws_size,
                              hipStream_t stream) {
    const float* x          = (const float*)d_in[0];
    const float* boundaries = (const float*)d_in[1];
    const float* in_proj_w  = (const float*)d_in[2];
    const float* in_proj_b  = (const float*)d_in[3];
    const float* out_w      = (const float*)d_in[4];
    const float* out_b      = (const float*)d_in[5];
    const float* w1         = (const float*)d_in[6];
    const float* b1         = (const float*)d_in[7];
    const float* w2         = (const float*)d_in[8];
    const float* b2         = (const float*)d_in[9];
    const float* ln_g       = (const float*)d_in[10];
    const float* ln_b       = (const float*)d_in[11];
    const float* pos_enc    = (const float*)d_in[12];
    const float* size_emb   = (const float*)d_in[13];
    float* out = (float*)d_out;

    char* ws = (char*)d_ws;
    auto al = [](size_t v) { return (v + 255) & ~(size_t)255; };

    // ---- workspace layout (bytes) ----
    const size_t sz_seg   = (size_t)Bb * Sq * 4;
    const size_t sz_start = (size_t)Bb * (Sq + 1) * 4;
    const size_t sz_ipw   = (size_t)3 * Dm * Dm * 2;
    const size_t sz_outw  = (size_t)Dm * Dm * 2;
    const size_t sz_w12   = (size_t)2 * Dm * Dm * 2;
    const size_t sz_xb    = (size_t)Bb * Sq * Dm * 2;
    const size_t sz_qkv   = (size_t)Bb * Sq * 3 * Dm * 2;
    const size_t sz_ctx   = (size_t)Bb * Sq * Dm * 2;
    const size_t sz_attn  = (size_t)Bb * Sq * Dm * 4;

    size_t off = 0;
    const size_t o_seg   = off; off = al(off + sz_seg);
    const size_t o_start = off; off = al(off + sz_start);
    const size_t o_nch   = off; off = al(off + Bb * 4);
    const size_t o_ipwb  = off; off = al(off + sz_ipw);
    const size_t o_outwb = off; off = al(off + sz_outw);
    const size_t o_w1b   = off; off = al(off + sz_w12);
    const size_t o_w2b   = off; off = al(off + sz_w12);
    const size_t o_xb    = off; off = al(off + sz_xb);
    const size_t o_qkv   = off; off = al(off + sz_qkv);
    const size_t o_ctx   = off; off = al(off + sz_ctx);
    const size_t o_attn  = off; off = al(off + sz_attn);

    int*            seg    = (int*)(ws + o_seg);
    int*            start  = (int*)(ws + o_start);
    int*            nch    = (int*)(ws + o_nch);
    unsigned short* ipwb   = (unsigned short*)(ws + o_ipwb);
    unsigned short* outwb  = (unsigned short*)(ws + o_outwb);
    unsigned short* w1b    = (unsigned short*)(ws + o_w1b);
    unsigned short* w2b    = (unsigned short*)(ws + o_w2b);
    unsigned short* xb     = (unsigned short*)(ws + o_xb);
    unsigned short* qkv    = (unsigned short*)(ws + o_qkv);
    unsigned short* ctx    = (unsigned short*)(ws + o_ctx);
    float*          attn   = (float*)(ws + o_attn);
    unsigned short* chunks = (unsigned short*)(ws + o_xb);   // alias: xb dead after QKV
    unsigned short* h1     = (unsigned short*)(ws + o_qkv);  // alias: qkv dead after attn
    float*          h2     = (float*)(ws + o_attn);          // alias: attn dead after pool

    const int M  = Bb * Sq;      // 8192
    const int Mc = Bb * MAXC;    // 2048

    // 0. bf16 pre-conversion of x and weights (bandwidth-bound)
    auto cvt = [&](const float* src, unsigned short* dst, size_t n) {
        int nq = (int)(n / 4);
        cvt_bf16<<<(nq + 255) / 256, 256, 0, stream>>>(src, dst, nq);
    };
    cvt(x,         xb,    (size_t)Bb * Sq * Dm);
    cvt(in_proj_w, ipwb,  (size_t)3 * Dm * Dm);
    cvt(out_w,     outwb, (size_t)Dm * Dm);
    cvt(w1,        w1b,   (size_t)2 * Dm * Dm);
    cvt(w2,        w2b,   (size_t)2 * Dm * Dm);

    // 1. segment structure
    seg_kernel<<<Bb, 32, 0, stream>>>(boundaries, seg, start, nch);

    // 2. QKV projection: [8192,1536] x [4608,1536]^T -> bf16
    gemm_bf16<<<dim3(3 * Dm / 64, M / 128), 128, 0, stream>>>(
        xb, ipwb, in_proj_b, qkv, M, 3 * Dm, Dm, 0, 1);

    // 3. per-chunk attention -> bf16 ctx
    attn_kernel<<<dim3(Sq / 16, Hh, Bb), 32, 0, stream>>>(qkv, seg, start, ctx);

    // 4. output projection: [8192,1536] x [1536,1536]^T -> fp32
    gemm_bf16<<<dim3(Dm / 64, M / 128), 128, 0, stream>>>(
        ctx, outwb, out_b, attn, M, Dm, Dm, 0, 0);

    // 5. ragged mean pool + size/pos embeddings -> bf16 chunks
    pool_kernel<<<dim3(MAXC, Bb), 256, 0, stream>>>(
        attn, start, nch, size_emb, pos_enc, chunks);

    // 6. MLP1 + exact GELU: [2048,1536] x [3072,1536]^T -> bf16
    gemm_bf16<<<dim3(2 * Dm / 64, Mc / 128), 128, 0, stream>>>(
        chunks, w1b, b1, h1, Mc, 2 * Dm, Dm, 1, 1);

    // 7. MLP2: [2048,3072] x [1536,3072]^T -> fp32
    gemm_bf16<<<dim3(Dm / 64, Mc / 128), 128, 0, stream>>>(
        h1, w2b, b2, h2, Mc, Dm, 2 * Dm, 0, 0);

    // 8. LayerNorm -> output [8,256,1536]
    ln_kernel<<<Mc, 256, 0, stream>>>(h2, ln_g, ln_b, out);
}
